// TransformerEncoder_6502580486404
// MI455X (gfx1250) — matrix operand order repaired and checked by reading, not compiled, not run on hardware
//
#include <hip/hip_runtime.h>
#include <hip/hip_bf16.h>

// ---------------------------------------------------------------------------
// TransformerEncoder forward (L=4, H=8, D=N=49, B=2048) for gfx1250 (MI455X).
// One workgroup (256 threads = 8 wave32) per batch element. All per-sample
// tensors live in LDS, padded 49 -> 64. Every matmul is a 64x64x64 product
// built from v_wmma_f32_16x16x32_f16 (f16 in, f32 accumulate): 16 C-tiles,
// 2 tiles per wave, 2 K-steps per tile.
//
// Weights are pre-packed ONCE per launch (fp32 -> padded 64x64 f16 "BT"
// tiles in d_ws); encoder blocks then stage them into LDS with 16B copies,
// using global_load_async_to_lds_b128 when the toolchain exposes it.
// ---------------------------------------------------------------------------

typedef __attribute__((ext_vector_type(16))) _Float16 v16h;
typedef __attribute__((ext_vector_type(8)))  _Float16 v8h;
typedef __attribute__((ext_vector_type(8)))  float    v8f;
typedef int v4i_async __attribute__((vector_size(16)));  // async-copy payload

#define DIM   49
#define PAD   64
#define NL    4
#define NH    8
#define NHID  25
#define NCLS  10
#define NMAT  168   // 5*32 per-head matrices + 2*4 MLP matrices
#define MELEM (PAD * PAD)

__device__ __forceinline__ v8f vzero8() { v8f z = {}; return z; }

// Fallback: stage a [49x49] fp32 weight W ([in=k][out=n]) into LDS as f16 in
// "BT" layout: dst[n*64 + k] = W[k*49 + n], zero-padded to 64x64.
__device__ __forceinline__ void stageW(_Float16* dst, const float* W) {
  __builtin_prefetch(W, 0, 3);   // global_prefetch_b8
  for (int p = threadIdx.x; p < MELEM; p += 256) {
    const int n = p >> 6;
    const int k = p & 63;
    float v = (n < DIM && k < DIM) ? W[k * DIM + n] : 0.0f;
    dst[p] = (_Float16)v;
  }
}

// Fast path: copy a pre-packed 64x64 f16 tile (8192 B) global -> LDS.
__device__ __forceinline__ void stageP(_Float16* dst, const _Float16* src) {
  const int t = threadIdx.x;
#if defined(__gfx1250__) && \
    __has_builtin(__builtin_amdgcn_global_load_async_to_lds_b128)
  // Async DMA into LDS: no VGPR round-trip, tracked by ASYNCcnt.
  __builtin_amdgcn_global_load_async_to_lds_b128(
      (v4i_async*)(src + t * 8), (v4i_async*)(dst + t * 8), 0, 0);
  __builtin_amdgcn_global_load_async_to_lds_b128(
      (v4i_async*)(src + 2048 + t * 8), (v4i_async*)(dst + 2048 + t * 8), 0, 0);
#if __has_builtin(__builtin_amdgcn_s_wait_asynccnt)
  __builtin_amdgcn_s_wait_asynccnt(0);
#else
  asm volatile("s_wait_asynccnt 0" ::: "memory");
#endif
#else
  v8h* d = (v8h*)dst;
  const v8h* s = (const v8h*)src;
  d[t] = s[t];             // global_load_b128 + ds_store_b128
  d[t + 256] = s[t + 256];
#endif
}

template <bool PACKED>
__device__ __forceinline__ void stage(_Float16* dst, const float* W,
                                      const _Float16* P, int midx) {
  if constexpr (PACKED) {
    stageP(dst, P + (size_t)midx * MELEM);
  } else {
    stageW(dst, W);
  }
}

// D(64x64) += A(64x64, f16 row-major) * B(64x64), B staged as BT[n*64+k].
// A-fragment (16-bit A 16x32, wave32): lanes 0-15 hold K {0..7,16..23},
// lanes 16-31 hold K {8..15,24..31} -> two contiguous v8h LDS loads.
// B-fragment: lanes 0-15 K {0..7,16..23}, lanes 16-31 K {8..15,24..31} for
// fixed N -> BT layout, two v8h LDS loads.
__device__ __forceinline__ void mm64_acc(const _Float16* A, const _Float16* BT,
                                         v8f c[2]) {
  const int lane = threadIdx.x & 31;
  const int wave = threadIdx.x >> 5;
  const int hi   = lane >> 4;
  const int ln   = lane & 15;
#pragma unroll
  for (int t = 0; t < 2; ++t) {
    const int tile = wave * 2 + t;
    const int ti = tile >> 2;
    const int tj = tile & 3;
#pragma unroll
    for (int kb = 0; kb < 2; ++kb) {
      const _Float16* ap = A + (ti * 16 + ln) * PAD + kb * 32 + hi * 8;
      v8h alo = *(const v8h*)(ap);
      v8h ahi = *(const v8h*)(ap + 16);
      v16h av = __builtin_shufflevector(alo, ahi, 0, 1, 2, 3, 4, 5, 6, 7,
                                        8, 9, 10, 11, 12, 13, 14, 15);
      const _Float16* bp = BT + (tj * 16 + ln) * PAD + kb * 32 + hi * 8;
      v8h blo = *(const v8h*)(bp);
      v8h bhi = *(const v8h*)(bp + 16);
      v16h bvv = __builtin_shufflevector(blo, bhi, 0, 1, 2, 3, 4, 5, 6, 7,
                                         8, 9, 10, 11, 12, 13, 14, 15);
      c[t] = __builtin_amdgcn_wmma_f32_16x16x32_f16(
          false, av, false, bvv, (short)0, c[t], false, false);
    }
  }
}

__device__ __forceinline__ void st16_rm(_Float16* D, const v8f c[2],
                                        const float* bias) {
  const int lane = threadIdx.x & 31;
  const int wave = threadIdx.x >> 5;
  const int hi = lane >> 4, ln = lane & 15;
#pragma unroll
  for (int t = 0; t < 2; ++t) {
    const int tile = wave * 2 + t;
    const int ti = tile >> 2, tj = tile & 3;
    const int n = tj * 16 + ln;
    float b = (bias && n < DIM) ? bias[n] : 0.0f;
#pragma unroll
    for (int r = 0; r < 8; ++r) {
      const int m = ti * 16 + r + hi * 8;
      D[m * PAD + n] = (_Float16)(c[t][r] + b);
    }
  }
}

__device__ __forceinline__ void st16_tr(_Float16* D, const v8f c[2],
                                        const float* bias) {
  const int lane = threadIdx.x & 31;
  const int wave = threadIdx.x >> 5;
  const int hi = lane >> 4, ln = lane & 15;
#pragma unroll
  for (int t = 0; t < 2; ++t) {
    const int tile = wave * 2 + t;
    const int ti = tile >> 2, tj = tile & 3;
    const int n = tj * 16 + ln;
    float b = (bias && n < DIM) ? bias[n] : 0.0f;
#pragma unroll
    for (int r = 0; r < 8; ++r) {
      const int m = ti * 16 + r + hi * 8;
      D[n * PAD + m] = (_Float16)(c[t][r] + b);
    }
  }
}

__device__ __forceinline__ void st32_rm(float* D, const v8f c[2]) {
  const int lane = threadIdx.x & 31;
  const int wave = threadIdx.x >> 5;
  const int hi = lane >> 4, ln = lane & 15;
#pragma unroll
  for (int t = 0; t < 2; ++t) {
    const int tile = wave * 2 + t;
    const int ti = tile >> 2, tj = tile & 3;
    const int n = tj * 16 + ln;
#pragma unroll
    for (int r = 0; r < 8; ++r) {
      const int m = ti * 16 + r + hi * 8;
      D[m * PAD + n] = c[t][r];
    }
  }
}

// Pre-pack [49x49] fp32 matrices into padded 64x64 f16 BT tiles.
__global__ __launch_bounds__(256) void pack_weights(
    const float* __restrict__ W, unsigned short* __restrict__ dstRaw) {
  const float* src = W + (size_t)blockIdx.x * DIM * DIM;
  _Float16* d = ((_Float16*)dstRaw) + (size_t)blockIdx.x * MELEM;
  for (int p = threadIdx.x; p < MELEM; p += 256) {
    const int n = p >> 6;
    const int k = p & 63;
    float v = (n < DIM && k < DIM) ? src[k * DIM + n] : 0.0f;
    d[p] = (_Float16)v;
  }
}

template <bool PACKED>
__global__ __launch_bounds__(256) void encoder_kernel(
    const float* __restrict__ emb, const int* __restrict__ labels,
    const float* __restrict__ Wq, const float* __restrict__ bq,
    const float* __restrict__ Wk, const float* __restrict__ bk,
    const float* __restrict__ Wv, const float* __restrict__ bv,
    const float* __restrict__ Wh, const float* __restrict__ bh,
    const float* __restrict__ Wout,
    const float* __restrict__ W1, const float* __restrict__ b1,
    const float* __restrict__ W2, const float* __restrict__ b2,
    const float* __restrict__ Wc1, const float* __restrict__ bc1,
    const float* __restrict__ Wc2, const float* __restrict__ bc2,
    const unsigned short* __restrict__ packRaw,
    float* __restrict__ out_loss, float* __restrict__ out_corr) {
  __shared__ __align__(16) _Float16 xs[MELEM];   // activation (A operand)
  __shared__ __align__(16) _Float16 wb[MELEM];   // staged weight (BT)
  __shared__ __align__(16) _Float16 q16[MELEM];  // Q / attn / head_out
  __shared__ __align__(16) _Float16 k16[MELEM];  // K (== BT of K^T) / ctx
  __shared__ __align__(16) _Float16 v16[MELEM];  // V transposed (BT)
  __shared__ __align__(16) float    sc[MELEM];   // f32 scores / scratch
  __shared__ float red[256];                     // softmax partials

  const int b = blockIdx.x;
  const int tid = threadIdx.x;
  const _Float16* packW = (const _Float16*)packRaw;
  constexpr float kInvScale = 1.0f / (7.0f + 1e-6f);  // 1/(sqrt(49)+eps)

  // Stage embedding -> xs (f16, zero padded).
  for (int p = tid; p < MELEM; p += 256) {
    const int r = p >> 6, cix = p & 63;
    float v = (r < DIM && cix < DIM) ? emb[(size_t)b * DIM * DIM + r * DIM + cix]
                                     : 0.0f;
    xs[p] = (_Float16)v;
  }
  __syncthreads();

  v8f c[2];
  for (int l = 0; l < NL; ++l) {
    v8f cacc[2];
    cacc[0] = vzero8(); cacc[1] = vzero8();
    for (int h = 0; h < NH; ++h) {
      const int lh = l * NH + h;
      const int wo = lh * DIM * DIM;
      const int bo = lh * DIM;

      // Q = x @ Wq + bq
      stage<PACKED>(wb, Wq + wo, packW, 0 + lh); __syncthreads();
      c[0] = vzero8(); c[1] = vzero8();
      mm64_acc(xs, wb, c); __syncthreads();
      st16_rm(q16, c, bq + bo); __syncthreads();

      // K = x @ Wk + bk   (row-major K == BT layout of K^T)
      stage<PACKED>(wb, Wk + wo, packW, 32 + lh); __syncthreads();
      c[0] = vzero8(); c[1] = vzero8();
      mm64_acc(xs, wb, c); __syncthreads();
      st16_rm(k16, c, bk + bo); __syncthreads();

      // V = x @ Wv + bv   (stored transposed -> BT layout)
      stage<PACKED>(wb, Wv + wo, packW, 64 + lh); __syncthreads();
      c[0] = vzero8(); c[1] = vzero8();
      mm64_acc(xs, wb, c); __syncthreads();
      st16_tr(v16, c, bv + bo); __syncthreads();

      // scores = Q @ K^T
      c[0] = vzero8(); c[1] = vzero8();
      mm64_acc(q16, k16, c); __syncthreads();
      st32_rm(sc, c); __syncthreads();

      // softmax(clip(scores/scale)) -> attn (f16) in q16; 4 threads per row.
      {
        const int row = tid >> 2, sub = tid & 3;
        float mx = -1e30f;
        for (int j = sub; j < DIM; j += 4) {
          float s = sc[row * PAD + j] * kInvScale;
          s = fminf(30.0f, fmaxf(-30.0f, s));
          sc[row * PAD + j] = s;
          mx = fmaxf(mx, s);
        }
        red[tid] = mx;
        __syncthreads();
        mx = fmaxf(fmaxf(red[row * 4 + 0], red[row * 4 + 1]),
                   fmaxf(red[row * 4 + 2], red[row * 4 + 3]));
        float sum = 0.0f;
        for (int j = sub; j < DIM; j += 4) {
          float e = __expf(sc[row * PAD + j] - mx);
          sc[row * PAD + j] = e;
          sum += e;
        }
        __syncthreads();
        red[tid] = sum;
        __syncthreads();
        sum = (red[row * 4 + 0] + red[row * 4 + 1]) +
              (red[row * 4 + 2] + red[row * 4 + 3]);
        const float inv = 1.0f / sum;
        for (int j = sub; j < DIM; j += 4)
          q16[row * PAD + j] = (_Float16)(sc[row * PAD + j] * inv);
        for (int j = DIM + sub; j < PAD; j += 4)
          q16[row * PAD + j] = (_Float16)0.0f;
      }
      __syncthreads();

      // ctx = attn @ V  -> k16
      c[0] = vzero8(); c[1] = vzero8();
      mm64_acc(q16, v16, c); __syncthreads();
      st16_rm(k16, c, nullptr); __syncthreads();

      // head_out = ctx @ Wh + bh  -> q16
      stage<PACKED>(wb, Wh + wo, packW, 96 + lh); __syncthreads();
      c[0] = vzero8(); c[1] = vzero8();
      mm64_acc(k16, wb, c); __syncthreads();
      st16_rm(q16, c, bh + bo); __syncthreads();

      // x_next += head_out @ Wout  (accumulate in registers across heads)
      stage<PACKED>(wb, Wout + wo, packW, 128 + lh); __syncthreads();
      mm64_acc(q16, wb, cacc); __syncthreads();
    }

    // x = sum over heads
    st16_rm(xs, cacc, nullptr); __syncthreads();

    // MLP between blocks (no activation): x = (x@W1+b1)@W2+b2
    stage<PACKED>(wb, W1 + l * DIM * DIM, packW, 160 + l); __syncthreads();
    c[0] = vzero8(); c[1] = vzero8();
    mm64_acc(xs, wb, c); __syncthreads();
    st16_rm(q16, c, b1 + l * DIM); __syncthreads();

    stage<PACKED>(wb, W2 + l * DIM * DIM, packW, 164 + l); __syncthreads();
    c[0] = vzero8(); c[1] = vzero8();
    mm64_acc(q16, wb, c); __syncthreads();
    st16_rm(xs, c, b2 + l * DIM); __syncthreads();
  }

  // pooled = mean over rows (patches) -> sc[0..48]
  if (tid < DIM) {
    float s = 0.0f;
    for (int n = 0; n < DIM; ++n) s += (float)xs[n * PAD + tid];
    sc[tid] = s * (1.0f / (float)DIM);
  }
  __syncthreads();
  if (tid < NHID) {
    float hsum = bc1[tid];
    for (int d = 0; d < DIM; ++d) hsum += sc[d] * Wc1[d * NHID + tid];
    sc[64 + tid] = hsum;
  }
  __syncthreads();
  if (tid < NCLS) {
    float z = bc2[tid];
    for (int j = 0; j < NHID; ++j) z += sc[64 + j] * Wc2[j * NCLS + tid];
    sc[128 + tid] = z;
  }
  __syncthreads();
  if (tid == 0) {
    float mx = -1e30f;
    int pred = 0;
    for (int k2 = 0; k2 < NCLS; ++k2) {
      if (sc[128 + k2] > mx) { mx = sc[128 + k2]; pred = k2; }
    }
    float sum = 0.0f;
    for (int k2 = 0; k2 < NCLS; ++k2) sum += __expf(sc[128 + k2] - mx);
    const float lse = mx + __logf(sum);
    const int lab = labels[b];
    out_loss[b] = lse - sc[128 + lab];
    out_corr[b] = (pred == lab) ? 1.0f : 0.0f;
  }
}

__global__ __launch_bounds__(256) void finalize_kernel(
    const float* __restrict__ per_loss, const float* __restrict__ per_corr,
    float* __restrict__ out, int Bn) {
  __shared__ float sl[256];
  __shared__ float sa[256];
  float a = 0.0f, cc = 0.0f;
  for (int i = threadIdx.x; i < Bn; i += 256) {
    a += per_loss[i];
    cc += per_corr[i];
  }
  sl[threadIdx.x] = a;
  sa[threadIdx.x] = cc;
  __syncthreads();
  for (int s = 128; s > 0; s >>= 1) {
    if (threadIdx.x < s) {
      sl[threadIdx.x] += sl[threadIdx.x + s];
      sa[threadIdx.x] += sa[threadIdx.x + s];
    }
    __syncthreads();
  }
  if (threadIdx.x == 0) {
    out[0] = sl[0] / (float)Bn;  // loss
    out[1] = sa[0] / (float)Bn;  // accuracy
  }
}

extern "C" void kernel_launch(void* const* d_in, const int* in_sizes, int n_in,
                              void* d_out, int out_size, void* d_ws,
                              size_t ws_size, hipStream_t stream) {
  const float* emb  = (const float*)d_in[0];
  const int* labels = (const int*)d_in[1];
  const float* Wq   = (const float*)d_in[2];
  const float* bq   = (const float*)d_in[3];
  const float* Wk   = (const float*)d_in[4];
  const float* bk   = (const float*)d_in[5];
  const float* Wv   = (const float*)d_in[6];
  const float* bv   = (const float*)d_in[7];
  const float* Wh   = (const float*)d_in[8];
  const float* bh   = (const float*)d_in[9];
  const float* Wout = (const float*)d_in[10];
  const float* W1   = (const float*)d_in[11];
  const float* b1   = (const float*)d_in[12];
  const float* W2   = (const float*)d_in[13];
  const float* b2   = (const float*)d_in[14];
  const float* Wc1  = (const float*)d_in[15];
  const float* bc1  = (const float*)d_in[16];
  const float* Wc2  = (const float*)d_in[17];
  const float* bc2  = (const float*)d_in[18];

  const int Bn = in_sizes[1];  // target_labels element count == batch
  float* wsl = (float*)d_ws;   // per-sample loss
  float* wsc = wsl + Bn;       // per-sample correct flag

  const size_t packOff = ((size_t)2 * Bn * sizeof(float) + 255) & ~(size_t)255;
  const size_t packBytes = (size_t)NMAT * MELEM * sizeof(unsigned short);
  const bool usePacked = (ws_size >= packOff + packBytes);

  if (usePacked) {
    unsigned short* pk = (unsigned short*)((char*)d_ws + packOff);
    pack_weights<<<NL * NH, 256, 0, stream>>>(Wq,   pk + (size_t)0   * MELEM);
    pack_weights<<<NL * NH, 256, 0, stream>>>(Wk,   pk + (size_t)32  * MELEM);
    pack_weights<<<NL * NH, 256, 0, stream>>>(Wv,   pk + (size_t)64  * MELEM);
    pack_weights<<<NL * NH, 256, 0, stream>>>(Wh,   pk + (size_t)96  * MELEM);
    pack_weights<<<NL * NH, 256, 0, stream>>>(Wout, pk + (size_t)128 * MELEM);
    pack_weights<<<NL, 256, 0, stream>>>(W1,        pk + (size_t)160 * MELEM);
    pack_weights<<<NL, 256, 0, stream>>>(W2,        pk + (size_t)164 * MELEM);
    encoder_kernel<true><<<Bn, 256, 0, stream>>>(
        emb, labels, Wq, bq, Wk, bk, Wv, bv, Wh, bh, Wout, W1, b1, W2, b2,
        Wc1, bc1, Wc2, bc2, pk, wsl, wsc);
  } else {
    encoder_kernel<false><<<Bn, 256, 0, stream>>>(
        emb, labels, Wq, bq, Wk, bk, Wv, bv, Wh, bh, Wout, W1, b1, W2, b2,
        Wc1, bc1, Wc2, bc2, nullptr, wsl, wsc);
  }
  finalize_kernel<<<1, 256, 0, stream>>>(wsl, wsc, (float*)d_out, Bn);
}
